// Mamba_channel_temporal_block_11836929868328
// MI455X (gfx1250) — matrix operand order repaired and checked by reading, not compiled, not run on hardware
//
#include <hip/hip_runtime.h>
#include <stdint.h>

#define D_MODEL 512
#define D_STATE 16
#define D_CONV  4
#define D_INNER 1024
#define DT_RANK 32
#define BATCH   4
#define SEQ     1024
#define NROWS   (BATCH * SEQ)   // 4096

typedef __attribute__((ext_vector_type(16))) __bf16 v16bf;
typedef __attribute__((ext_vector_type(8)))  __bf16 v8bf;
typedef __attribute__((ext_vector_type(8)))  float  v8f;

__device__ __forceinline__ uint16_t f32_to_bf16_bits(float f) {
    union { float f; uint32_t u; } v; v.f = f;
    uint32_t u = v.u;
    uint32_t lsb = (u >> 16) & 1u;
    u += 0x7fffu + lsb;               // round to nearest even
    return (uint16_t)(u >> 16);
}

// ---- WMMA fragment loaders (wave32; this code's contiguous k order, same for A and B) ----------
// A (16x32 bf16): lane l<16 holds row M=l, K = 0..15; lane l>=16 holds
// row M=l-16, K = 16..31.  One contiguous 32B load.
__device__ __forceinline__ v16bf load_a_frag(const uint16_t* Arow, int k0, int half) {
    return *(const v16bf*)(Arow + k0 + half * 16);
}
// B (32x16 bf16): lane n<16 holds column N=n, K=0..15; lane n>=16 holds column
// N=n-16, K=16..31.  One contiguous 32B load from row-major weight W[N][K].
__device__ __forceinline__ v16bf load_b_frag(const uint16_t* p) {
    return *(const v16bf*)p;
}

// ---------------------------------------------------------------------------
// 1) LayerNorm: feature[4096,512] -> xn (f32 for residual) + xn (bf16 for GEMM)
__global__ void k_layernorm(const float* __restrict__ feat,
                            const float* __restrict__ w,
                            const float* __restrict__ b,
                            float* __restrict__ xn_f32,
                            uint16_t* __restrict__ xn_bf16) {
    int row = blockIdx.x;
    int tid = threadIdx.x;                 // 256 threads, 2 elems each
    const float* x = feat + row * D_MODEL;
    float v0 = x[tid], v1 = x[tid + 256];
    __shared__ float rs[256], rq[256];
    rs[tid] = v0 + v1;
    rq[tid] = v0 * v0 + v1 * v1;
    __syncthreads();
    for (int off = 128; off > 0; off >>= 1) {
        if (tid < off) { rs[tid] += rs[tid + off]; rq[tid] += rq[tid + off]; }
        __syncthreads();
    }
    float mu  = rs[0] * (1.0f / 512.0f);
    float var = rq[0] * (1.0f / 512.0f) - mu * mu;
    float inv = rsqrtf(var + 1e-5f);
    float y0 = (v0 - mu) * inv * w[tid]       + b[tid];
    float y1 = (v1 - mu) * inv * w[tid + 256] + b[tid + 256];
    xn_f32[row * D_MODEL + tid]        = y0;
    xn_f32[row * D_MODEL + tid + 256]  = y1;
    xn_bf16[row * D_MODEL + tid]       = f32_to_bf16_bits(y0);
    xn_bf16[row * D_MODEL + tid + 256] = f32_to_bf16_bits(y1);
}

// 2) generic f32 -> bf16 weight convert
__global__ void k_f32_to_bf16(const float* __restrict__ src,
                              uint16_t* __restrict__ dst, int n) {
    int i = blockIdx.x * blockDim.x + threadIdx.x;
    if (i < n) dst[i] = f32_to_bf16_bits(src[i]);
}

// 3) in_proj: xz[m,e] = sum_d xn[m,d] * Win[e,d]; split halves into x / z bufs.
//    One wave per 16x64 strip. grid = (M/16, 2048/64)
__global__ void k_gemm_inproj(const uint16_t* __restrict__ A,   // [4096,512] bf16
                              const uint16_t* __restrict__ W,   // [2048,512] bf16
                              float* __restrict__ xout,         // [4096,1024]
                              float* __restrict__ zout) {       // [4096,1024]
    const int K = D_MODEL;
    int m0 = blockIdx.x * 16;
    int n0 = blockIdx.y * 64;
    int lane = threadIdx.x & 31;
    int r = lane & 15, half = lane >> 4;
    const uint16_t* Arow = A + (m0 + r) * K;
    v8f acc[4] = {{}, {}, {}, {}};
    for (int k0 = 0; k0 < K; k0 += 32) {
        v16bf a = load_a_frag(Arow, k0, half);
#pragma unroll
        for (int t = 0; t < 4; ++t) {
            v16bf bf = load_b_frag(W + (n0 + t * 16 + r) * K + k0 + half * 16);
            acc[t] = __builtin_amdgcn_wmma_f32_16x16x32_bf16(
                false, a, false, bf, (short)0, acc[t], false, false);
        }
    }
    int mbase = m0 + half * 8;   // D layout: VGPR i -> M = mbase+i, N = n-tile + r
#pragma unroll
    for (int t = 0; t < 4; ++t) {
        int e = n0 + t * 16 + r;
        float* dst = (e < D_INNER) ? xout : zout;
        int col = (e < D_INNER) ? e : (e - D_INNER);
#pragma unroll
        for (int i = 0; i < 8; ++i)
            dst[(mbase + i) * D_INNER + col] = acc[t][i];
    }
}

// 4) causal depthwise conv (K=4) + SiLU; in-place on xbuf, plus bf16 copy.
__global__ void k_conv_silu(float* __restrict__ xbuf,           // [4096,1024] in/out
                            const float* __restrict__ cw,       // [1024,4]
                            const float* __restrict__ cb,       // [1024]
                            uint16_t* __restrict__ x_bf16) {    // [4096,1024]
    int t = blockIdx.x * blockDim.x + threadIdx.x;  // 4096 = B * D_INNER
    int d = t & (D_INNER - 1), b = t >> 10;
    float w0 = cw[d * 4 + 0], w1 = cw[d * 4 + 1];
    float w2 = cw[d * 4 + 2], w3 = cw[d * 4 + 3];
    float bias = cb[d];
    float x0 = 0.f, x1 = 0.f, x2 = 0.f;
    for (int l = 0; l < SEQ; ++l) {
        int idx = (b * SEQ + l) * D_INNER + d;
        float x3 = xbuf[idx];
        float acc = w0 * x0 + w1 * x1 + w2 * x2 + w3 * x3 + bias;
        float out = acc / (1.0f + __expf(-acc));   // SiLU
        xbuf[idx] = out;
        x_bf16[idx] = f32_to_bf16_bits(out);
        x0 = x1; x1 = x2; x2 = x3;
    }
}

// 5) x_proj: x_dbl[m,j] = sum_d x[m,d] * Wxp[j,d], j<64.  grid = (M/16)
__global__ void k_gemm_xproj(const uint16_t* __restrict__ A,     // [4096,1024] bf16
                             const uint16_t* __restrict__ W,     // [64,1024] bf16
                             float* __restrict__ xdbl,           // [4096,64]
                             uint16_t* __restrict__ xdbl_bf16) { // [4096,32] (dt ranks)
    const int K = D_INNER;
    int m0 = blockIdx.x * 16;
    int lane = threadIdx.x & 31;
    int r = lane & 15, half = lane >> 4;
    const uint16_t* Arow = A + (m0 + r) * K;
    v8f acc[4] = {{}, {}, {}, {}};
    for (int k0 = 0; k0 < K; k0 += 32) {
        v16bf a = load_a_frag(Arow, k0, half);
#pragma unroll
        for (int t = 0; t < 4; ++t) {
            v16bf bf = load_b_frag(W + (t * 16 + r) * K + k0 + half * 16);
            acc[t] = __builtin_amdgcn_wmma_f32_16x16x32_bf16(
                false, a, false, bf, (short)0, acc[t], false, false);
        }
    }
    int mbase = m0 + half * 8;
#pragma unroll
    for (int t = 0; t < 4; ++t) {
        int e = t * 16 + r;
#pragma unroll
        for (int i = 0; i < 8; ++i) {
            float v = acc[t][i];
            int m = mbase + i;
            xdbl[m * 64 + e] = v;
            if (e < DT_RANK) xdbl_bf16[m * DT_RANK + e] = f32_to_bf16_bits(v);
        }
    }
}

// 6) dt_proj (K=32, single WMMA step) + bias + softplus.  grid = (M/16, 1024/64)
__global__ void k_gemm_dtproj(const uint16_t* __restrict__ A,   // [4096,32] bf16
                              const uint16_t* __restrict__ W,   // [1024,32] bf16
                              const float* __restrict__ bias,   // [1024]
                              float* __restrict__ dt) {         // [4096,1024]
    const int K = DT_RANK;
    int m0 = blockIdx.x * 16;
    int n0 = blockIdx.y * 64;
    int lane = threadIdx.x & 31;
    int r = lane & 15, half = lane >> 4;
    v16bf a = load_a_frag(A + (m0 + r) * K, 0, half);
    v8f acc[4] = {{}, {}, {}, {}};
#pragma unroll
    for (int t = 0; t < 4; ++t) {
        v16bf bf = load_b_frag(W + (n0 + t * 16 + r) * K + half * 16);
        acc[t] = __builtin_amdgcn_wmma_f32_16x16x32_bf16(
            false, a, false, bf, (short)0, acc[t], false, false);
    }
    int mbase = m0 + half * 8;
#pragma unroll
    for (int t = 0; t < 4; ++t) {
        int col = n0 + t * 16 + r;
#pragma unroll
        for (int i = 0; i < 8; ++i) {
            float v = acc[t][i] + bias[col];
            float sp = (v > 20.0f) ? v : log1pf(__expf(v));   // softplus
            dt[(mbase + i) * D_INNER + col] = sp;
        }
    }
}

// 7) selective scan: one lane per (b, d, s); 16-lane butterfly reduces over s.
//    Fuses +x*D, SiLU(z) gating, and bf16 downconvert for out_proj.
__global__ void k_scan(const float* __restrict__ dt,        // [4096,1024]
                       const float* __restrict__ x,         // [4096,1024] (post conv/silu)
                       const float* __restrict__ xdbl,      // [4096,64] (B at 32.., C at 48..)
                       const float* __restrict__ z,         // [4096,1024]
                       const float* __restrict__ A_log,     // [1024,16]
                       const float* __restrict__ Dp,        // [1024]
                       uint16_t* __restrict__ y_bf16) {     // [4096,1024]
    int t = blockIdx.x * blockDim.x + threadIdx.x;          // 65536 threads
    int s = t & (D_STATE - 1);
    int g = t >> 4;
    int d = g & (D_INNER - 1), b = g >> 10;
    float As = -__expf(A_log[d * D_STATE + s]);
    float Dd = Dp[d];
    float h = 0.0f;
    for (int l = 0; l < SEQ; ++l) {
        int m = b * SEQ + l;
        float dtv = dt[m * D_INNER + d];
        float xv  = x[m * D_INNER + d];
        float Bv  = xdbl[m * 64 + DT_RANK + s];
        float Cv  = xdbl[m * 64 + DT_RANK + D_STATE + s];
        h = __expf(dtv * As) * h + (dtv * xv) * Bv;
        float p = h * Cv;
        p += __shfl_xor(p, 1, 16);
        p += __shfl_xor(p, 2, 16);
        p += __shfl_xor(p, 4, 16);
        p += __shfl_xor(p, 8, 16);
        if (s == 0) {
            float y = p + xv * Dd;
            float zv = z[m * D_INNER + d];
            y *= zv / (1.0f + __expf(-zv));                  // * SiLU(z)
            y_bf16[m * D_INNER + d] = f32_to_bf16_bits(y);
        }
    }
}

// 8) out_proj + residual: out[m,e] = xn[m,e] + sum_d y[m,d]*Wout[e,d]
//    grid = (M/16, 512/64)
__global__ void k_gemm_outproj(const uint16_t* __restrict__ A,   // [4096,1024] bf16
                               const uint16_t* __restrict__ W,   // [512,1024] bf16
                               const float* __restrict__ xn,     // [4096,512]
                               float* __restrict__ out) {        // [4096,512]
    const int K = D_INNER;
    int m0 = blockIdx.x * 16;
    int n0 = blockIdx.y * 64;
    int lane = threadIdx.x & 31;
    int r = lane & 15, half = lane >> 4;
    const uint16_t* Arow = A + (m0 + r) * K;
    v8f acc[4] = {{}, {}, {}, {}};
    for (int k0 = 0; k0 < K; k0 += 32) {
        v16bf a = load_a_frag(Arow, k0, half);
#pragma unroll
        for (int t = 0; t < 4; ++t) {
            v16bf bf = load_b_frag(W + (n0 + t * 16 + r) * K + k0 + half * 16);
            acc[t] = __builtin_amdgcn_wmma_f32_16x16x32_bf16(
                false, a, false, bf, (short)0, acc[t], false, false);
        }
    }
    int mbase = m0 + half * 8;
#pragma unroll
    for (int t = 0; t < 4; ++t) {
        int e = n0 + t * 16 + r;
#pragma unroll
        for (int i = 0; i < 8; ++i) {
            int m = mbase + i;
            out[m * D_MODEL + e] = xn[m * D_MODEL + e] + acc[t][i];
        }
    }
}

// ---------------------------------------------------------------------------
extern "C" void kernel_launch(void* const* d_in, const int* in_sizes, int n_in,
                              void* d_out, int out_size, void* d_ws, size_t ws_size,
                              hipStream_t stream) {
    const float* feat      = (const float*)d_in[0];
    const float* ln_w      = (const float*)d_in[1];
    const float* ln_b      = (const float*)d_in[2];
    const float* in_proj_w = (const float*)d_in[3];
    const float* conv_w    = (const float*)d_in[4];
    const float* conv_b    = (const float*)d_in[5];
    const float* x_proj_w  = (const float*)d_in[6];
    const float* dt_proj_w = (const float*)d_in[7];
    const float* dt_proj_b = (const float*)d_in[8];
    const float* A_log     = (const float*)d_in[9];
    const float* Dp        = (const float*)d_in[10];
    const float* out_proj_w= (const float*)d_in[11];

    char* ws = (char*)d_ws;
    size_t off = 0;
    auto alloc = [&](size_t bytes) {
        void* p = ws + off;
        off += (bytes + 255) & ~(size_t)255;
        return p;
    };
    float*    xn_f32    = (float*)   alloc((size_t)NROWS * D_MODEL * 4);   //  8 MB
    uint16_t* xn_bf16   = (uint16_t*)alloc((size_t)NROWS * D_MODEL * 2);   //  4 MB
    uint16_t* w_in_bf   = (uint16_t*)alloc((size_t)2 * D_INNER * D_MODEL * 2); // 2 MB
    uint16_t* w_xp_bf   = (uint16_t*)alloc((size_t)64 * D_INNER * 2);      // 128 KB
    uint16_t* w_dt_bf   = (uint16_t*)alloc((size_t)D_INNER * DT_RANK * 2); //  64 KB
    uint16_t* w_out_bf  = (uint16_t*)alloc((size_t)D_MODEL * D_INNER * 2); //  1 MB
    float*    xbuf      = (float*)   alloc((size_t)NROWS * D_INNER * 4);   // 16 MB
    float*    zbuf      = (float*)   alloc((size_t)NROWS * D_INNER * 4);   // 16 MB
    uint16_t* xc_bf16   = (uint16_t*)alloc((size_t)NROWS * D_INNER * 2);   //  8 MB
    float*    xdbl      = (float*)   alloc((size_t)NROWS * 64 * 4);        //  1 MB
    uint16_t* xdbl_bf16 = (uint16_t*)alloc((size_t)NROWS * DT_RANK * 2);   // 256 KB
    float*    dtbuf     = (float*)   alloc((size_t)NROWS * D_INNER * 4);   // 16 MB
    uint16_t* y_bf16    = (uint16_t*)alloc((size_t)NROWS * D_INNER * 2);   //  8 MB

    // 1) LayerNorm
    k_layernorm<<<NROWS, 256, 0, stream>>>(feat, ln_w, ln_b, xn_f32, xn_bf16);

    // 2) weight conversions to bf16
    k_f32_to_bf16<<<(2 * D_INNER * D_MODEL + 255) / 256, 256, 0, stream>>>(in_proj_w, w_in_bf, 2 * D_INNER * D_MODEL);
    k_f32_to_bf16<<<(64 * D_INNER + 255) / 256, 256, 0, stream>>>(x_proj_w, w_xp_bf, 64 * D_INNER);
    k_f32_to_bf16<<<(D_INNER * DT_RANK + 255) / 256, 256, 0, stream>>>(dt_proj_w, w_dt_bf, D_INNER * DT_RANK);
    k_f32_to_bf16<<<(D_MODEL * D_INNER + 255) / 256, 256, 0, stream>>>(out_proj_w, w_out_bf, D_MODEL * D_INNER);

    // 3) in_proj GEMM (WMMA), split x/z
    {
        dim3 g(NROWS / 16, (2 * D_INNER) / 64);
        k_gemm_inproj<<<g, 32, 0, stream>>>(xn_bf16, w_in_bf, xbuf, zbuf);
    }

    // 4) causal conv + SiLU (in-place) + bf16 copy
    k_conv_silu<<<(BATCH * D_INNER) / 256, 256, 0, stream>>>(xbuf, conv_w, conv_b, xc_bf16);

    // 5) x_proj GEMM (WMMA)
    k_gemm_xproj<<<NROWS / 16, 32, 0, stream>>>(xc_bf16, w_xp_bf, xdbl, xdbl_bf16);

    // 6) dt_proj GEMM (WMMA) + softplus
    {
        dim3 g(NROWS / 16, D_INNER / 64);
        k_gemm_dtproj<<<g, 32, 0, stream>>>(xdbl_bf16, w_dt_bf, dt_proj_b, dtbuf);
    }

    // 7) selective scan + gating
    k_scan<<<(NROWS * D_STATE) / 256, 256, 0, stream>>>(dtbuf, xbuf, xdbl, zbuf, A_log, Dp, y_bf16);

    // 8) out_proj GEMM (WMMA) + residual -> d_out
    {
        dim3 g(NROWS / 16, D_MODEL / 64);
        k_gemm_outproj<<<g, 32, 0, stream>>>(y_bf16, w_out_bf, xn_f32, (float*)d_out);
    }
}
